// Agent_77979426226837
// MI455X (gfx1250) — compile-verified, matrix operand order checked by reading, not run on hardware
//
#include <hip/hip_runtime.h>
#include <hip/hip_bf16.h>

typedef __attribute__((ext_vector_type(16))) __bf16 v16bf;
typedef __attribute__((ext_vector_type(8)))  float  v8f;
typedef __attribute__((__vector_size__(16))) int    v4i;

#ifndef __has_builtin
#define __has_builtin(x) 0
#endif
#if __has_builtin(__builtin_amdgcn_global_load_async_to_lds_b128)
#define HAVE_ASYNC 1
#else
#define HAVE_ASYNC 0
#endif

#define AS1 __attribute__((address_space(1)))
#define AS3 __attribute__((address_space(3)))

#define GAMMA   0.99f
#define LAMBDA  0.95f
#define NU      0.001f
#define LOG_2PI 1.8378770664093453f

#define MB  256    // batch
#define TT  64     // time
#define HH  512
#define ZF  1024   // R*C
#define AA  64
#define DD  1536   // H + R*C
#define H1  1024
#define H2  1024
#define MROWS (MB*TT)   // 16384

// GEMM tiling
#define BM 128
#define BN 128
#define BK 32
#define LP 40      // LDS row pitch in bf16 elems (80B, 16B-aligned, bank-skewed)

// ---------- helpers ----------
static __device__ __forceinline__ unsigned short f2bfbits(float x) {
    unsigned u = __builtin_bit_cast(unsigned, x);
    return (unsigned short)((u + 0x7FFFu + ((u >> 16) & 1u)) >> 16);
}
static __device__ __forceinline__ float bf2f(unsigned short s) {
    unsigned u = ((unsigned)s) << 16;
    return __builtin_bit_cast(float, u);
}

// ---------- pack X0 = bf16 concat(h, z)  [MROWS x DD] ----------
__global__ __launch_bounds__(256) void Agent_pack_x0(
    const float* __restrict__ h, const float* __restrict__ z,
    unsigned short* __restrict__ X0)
{
    const int row = blockIdx.x;
    const float* hr = h + (size_t)row * HH;
    const float* zr = z + (size_t)row * ZF;
    unsigned short* o = X0 + (size_t)row * DD;
    for (int c = threadIdx.x; c < HH; c += 256) o[c]      = f2bfbits(hr[c]);
    for (int c = threadIdx.x; c < ZF; c += 256) o[HH + c] = f2bfbits(zr[c]);
}

// ---------- transpose+convert weights: WT[n][k] = bf16(W[k][n]) ----------
__global__ __launch_bounds__(256) void Agent_transpose_w(
    const float* __restrict__ W, unsigned short* __restrict__ WT, int K, int N)
{
    __shared__ float tile[32][33];
    const int n0 = blockIdx.x * 32, k0 = blockIdx.y * 32;
    const int tx = threadIdx.x & 31, ty = threadIdx.x >> 5;   // 32 x 8
#pragma unroll
    for (int i = ty; i < 32; i += 8) tile[i][tx] = W[(size_t)(k0 + i) * N + n0 + tx];
    __syncthreads();
#pragma unroll
    for (int i = ty; i < 32; i += 8)
        WT[(size_t)(n0 + i) * K + k0 + tx] = f2bfbits(tile[tx][i]);
}

// ---------- bf16 WMMA GEMM + bias + SiLU ----------
// A: bf16 [M x K] row-major; WT: bf16 [N x K] (transposed weights); Out bf16 [M x N].
// Block tile 128x128, 8 waves (4x2), wave patch 32x64 = 2x4 wmma tiles.
// Double-buffered LDS staged with GLOBAL_LOAD_ASYNC_TO_LDS_B128 when available.
__global__ __launch_bounds__(256) void Agent_gemm_bf16_silu(
    const unsigned short* __restrict__ A, const unsigned short* __restrict__ WT,
    const float* __restrict__ bias, unsigned short* __restrict__ Out,
    int K, int N)
{
    __shared__ unsigned short sA[2][BM][LP];   // [m][k]
    __shared__ unsigned short sB[2][BN][LP];   // [n][k]

    const int tid  = threadIdx.x;
    const int bm   = blockIdx.y * BM;
    const int bn   = blockIdx.x * BN;
    const int wave = tid >> 5;
    const int lane = tid & 31;
    const int half = lane >> 4;
    const int lr   = lane & 15;
    const int wm   = (wave & 3) * 32;   // 4 waves along M
    const int wn   = (wave >> 2) * 64;  // 2 waves along N

    // per-lane K indices for 16-bit fragments (ISA 7.12.2, 16-bit A 16x32)
    int ko[16];
#pragma unroll
    for (int i = 0; i < 16; ++i) ko[i] = i + 8 * half + ((i >= 8) ? 8 : 0);

    v8f acc[2][4];
#pragma unroll
    for (int mi = 0; mi < 2; ++mi)
#pragma unroll
        for (int ni = 0; ni < 4; ++ni)
#pragma unroll
            for (int j = 0; j < 8; ++j) acc[mi][ni][j] = 0.0f;

    // Stage one K-slice (BK) of A and B tiles into LDS buffer `st`.
    // 128 rows x 32 k bf16 = 512 x 16B chunks per tile; 2 chunks/thread/tile.
    auto stage = [&](int st, int k0) {
#pragma unroll
        for (int j = 0; j < 2; ++j) {
            const int cid = tid + 256 * j;
            const int row = cid >> 2;          // 0..127
            const int seg = (cid & 3) * 8;     // bf16 offset of 16B chunk
            const unsigned short* gA = A  + (size_t)(bm + row) * K + k0 + seg;
            const unsigned short* gB = WT + (size_t)(bn + row) * K + k0 + seg;
            unsigned short* lA = &sA[st][row][seg];
            unsigned short* lB = &sB[st][row][seg];
#if HAVE_ASYNC
            __builtin_amdgcn_global_load_async_to_lds_b128((AS1 v4i*)gA, (AS3 v4i*)lA, 0, 0);
            __builtin_amdgcn_global_load_async_to_lds_b128((AS1 v4i*)gB, (AS3 v4i*)lB, 0, 0);
#else
            *(uint4*)lA = *(const uint4*)gA;
            *(uint4*)lB = *(const uint4*)gB;
#endif
        }
    };

    const int nk = K >> 5;
    stage(0, 0);
    for (int kk = 0; kk < nk; ++kk) {
#if HAVE_ASYNC
        asm volatile("s_wait_asynccnt 0" ::: "memory");
#endif
        __syncthreads();   // stage kk visible to all; all reads of buf (kk+1)&1 done
        if (kk + 1 < nk) stage((kk + 1) & 1, (kk + 1) << 5);

        const int st = kk & 1;
        v16bf aF[2], bF[4];
#pragma unroll
        for (int mi = 0; mi < 2; ++mi) {
            const unsigned short* pr = sA[st][wm + mi * 16 + lr];
#pragma unroll
            for (int i = 0; i < 16; ++i)
                aF[mi][i] = __builtin_bit_cast(__bf16, pr[ko[i]]);
        }
#pragma unroll
        for (int ni = 0; ni < 4; ++ni) {
            const unsigned short* pc = sB[st][wn + ni * 16 + lr];
#pragma unroll
            for (int i = 0; i < 16; ++i)
                bF[ni][i] = __builtin_bit_cast(__bf16, pc[ko[i]]);
        }
#pragma unroll
        for (int mi = 0; mi < 2; ++mi)
#pragma unroll
            for (int ni = 0; ni < 4; ++ni)
                acc[mi][ni] = __builtin_amdgcn_wmma_f32_16x16x32_bf16(
                    false, aF[mi], false, bF[ni], (short)0, acc[mi][ni],
                    false, false);
    }

    // epilogue: bias + SiLU -> bf16.  D layout: lane lr -> N, vgpr j -> M=j+8*half
#pragma unroll
    for (int ni = 0; ni < 4; ++ni) {
        const int col = bn + wn + ni * 16 + lr;
        const float bv = bias[col];
#pragma unroll
        for (int mi = 0; mi < 2; ++mi) {
#pragma unroll
            for (int j = 0; j < 8; ++j) {
                const int row = bm + wm + mi * 16 + j + 8 * half;
                float x = acc[mi][ni][j] + bv;
                float s = x / (1.0f + expf(-x));   // SiLU
                Out[(size_t)row * N + col] = f2bfbits(s);
            }
        }
    }
}

// ---------- value = X2 @ W3 + b3 (wave per row) ----------
__global__ __launch_bounds__(256) void Agent_matvec(
    const unsigned short* __restrict__ X2, const float* __restrict__ W3,
    const float* __restrict__ b3, float* __restrict__ value, int K)
{
    const int row  = blockIdx.x * 8 + (threadIdx.x >> 5);
    const int lane = threadIdx.x & 31;
    const unsigned short* xr = X2 + (size_t)row * K;
    float s = 0.0f;
    for (int k = lane; k < K; k += 32) s += bf2f(xr[k]) * W3[k];
#pragma unroll
    for (int off = 16; off > 0; off >>= 1) s += __shfl_xor(s, off, 32);
    if (lane == 0) value[row] = s + b3[0];
}

// ---------- TD(lambda) scan + actor/critic partial losses ----------
__global__ __launch_bounds__(64) void Agent_scan_loss(
    const float* __restrict__ value, const float* __restrict__ reward,
    const float* __restrict__ cont, const float* __restrict__ action,
    const float* __restrict__ a_mu, const float* __restrict__ a_sigma,
    float* __restrict__ partials)
{
    __shared__ float sv[TT], sR[TT], ra[TT], rc[TT];
    const int b = blockIdx.x, t = threadIdx.x;
    const int m = b * TT + t;
    const float v = value[m];
    sv[t] = v;
    __syncthreads();

    if (t == 0) {  // backward TD(lambda) recursion, tiny serial loop
        float Rn = sv[TT - 1];
        sR[TT - 1] = Rn;
        for (int i = TT - 2; i >= 0; --i) {
            const float r = reward[b * TT + i];
            const float c = cont[b * TT + i];
            const float Rc = r + GAMMA * c * ((1.0f - LAMBDA) * sv[i + 1] + LAMBDA * Rn);
            sR[i] = Rc; Rn = Rc;
        }
    }
    __syncthreads();

    const float R   = sR[t];
    const float adv = R - v;
    float actor = 0.0f;
    const size_t base = (size_t)m * AA;
#pragma unroll 4
    for (int a = 0; a < AA; ++a) {
        const float sg = a_sigma[base + a];
        const float ls = logf(sg);
        const float d  = (action[base + a] - a_mu[base + a]) / sg;
        const float lp = -0.5f * d * d - ls - 0.5f * LOG_2PI;
        const float en = 0.5f + 0.5f * LOG_2PI + ls;
        actor += lp * adv + NU * en;
    }
    ra[t] = actor;
    rc[t] = 0.5f * adv * adv;
    __syncthreads();
    for (int s = 32; s > 0; s >>= 1) {
        if (t < s) { ra[t] += ra[t + s]; rc[t] += rc[t + s]; }
        __syncthreads();
    }
    if (t == 0) { partials[b * 2 + 0] = ra[0]; partials[b * 2 + 1] = rc[0]; }
}

// ---------- finalize ----------
__global__ __launch_bounds__(256) void Agent_finalize(
    const float* __restrict__ partials, float* __restrict__ out)
{
    __shared__ float sa[MB], sc[MB];
    const int t = threadIdx.x;
    sa[t] = partials[t * 2 + 0];
    sc[t] = partials[t * 2 + 1];
    __syncthreads();
    for (int s = 128; s > 0; s >>= 1) {
        if (t < s) { sa[t] += sa[t + s]; sc[t] += sc[t + s]; }
        __syncthreads();
    }
    if (t == 0) {
        out[0] = -sa[0] / (float)(MB * TT * AA);
        out[1] =  sc[0] / (float)(MB * TT);
    }
}

extern "C" void kernel_launch(void* const* d_in, const int* in_sizes, int n_in,
                              void* d_out, int out_size, void* d_ws, size_t ws_size,
                              hipStream_t stream) {
    (void)in_sizes; (void)n_in; (void)out_size; (void)ws_size;
    const float* h       = (const float*)d_in[0];
    const float* z       = (const float*)d_in[1];
    const float* reward  = (const float*)d_in[2];
    const float* cont    = (const float*)d_in[3];
    const float* action  = (const float*)d_in[4];
    const float* a_mu    = (const float*)d_in[5];
    const float* a_sigma = (const float*)d_in[6];
    const float* W1      = (const float*)d_in[7];
    const float* b1      = (const float*)d_in[8];
    const float* W2      = (const float*)d_in[9];
    const float* b2      = (const float*)d_in[10];
    const float* W3      = (const float*)d_in[11];
    const float* b3      = (const float*)d_in[12];
    float* out = (float*)d_out;

    unsigned short* X0  = (unsigned short*)d_ws;            // [16384 x 1536]
    unsigned short* WT1 = X0  + (size_t)MROWS * DD;         // [1024 x 1536]
    unsigned short* WT2 = WT1 + (size_t)H1 * DD;            // [1024 x 1024]
    unsigned short* X1  = WT2 + (size_t)H2 * H1;            // [16384 x 1024]
    unsigned short* X2  = X1  + (size_t)MROWS * H1;         // [16384 x 1024]
    float* value    = (float*)(X2 + (size_t)MROWS * H2);
    float* partials = value + MROWS;

    dim3 blk(256);
    // pre-pack activations + weights to bf16 (weights transposed to [N][K])
    Agent_pack_x0<<<dim3(MROWS), blk, 0, stream>>>(h, z, X0);
    Agent_transpose_w<<<dim3(H1 / 32, DD / 32), blk, 0, stream>>>(W1, WT1, DD, H1);
    Agent_transpose_w<<<dim3(H2 / 32, H1 / 32), blk, 0, stream>>>(W2, WT2, H1, H2);
    // layer 1: [16384 x 1536] @ [1536 x 1024] (+b1, SiLU) -> X1
    Agent_gemm_bf16_silu<<<dim3(H1 / BN, MROWS / BM), blk, 0, stream>>>(
        X0, WT1, b1, X1, DD, H1);
    // layer 2: [16384 x 1024] @ [1024 x 1024] (+b2, SiLU) -> X2
    Agent_gemm_bf16_silu<<<dim3(H2 / BN, MROWS / BM), blk, 0, stream>>>(
        X1, WT2, b2, X2, H1, H2);
    // value head
    Agent_matvec<<<dim3(MROWS / 8), blk, 0, stream>>>(X2, W3, b3, value, H2);
    // TD(lambda) + losses
    Agent_scan_loss<<<dim3(MB), dim3(TT), 0, stream>>>(
        value, reward, cont, action, a_mu, a_sigma, partials);
    Agent_finalize<<<dim3(1), dim3(256), 0, stream>>>(partials, out);
}
